// BandRoFormerEncoderLayer_61022895342184
// MI455X (gfx1250) — matrix operand order repaired and checked by reading, not compiled, not run on hardware
//
#include <hip/hip_runtime.h>
#include <math.h>

#define DIM   512
#define HEADS 8
#define HD    64
#define BATCH 4
#define SEQ   2048
#define NTOK  (BATCH*SEQ)

typedef __attribute__((ext_vector_type(16))) _Float16 v16h;
typedef __attribute__((ext_vector_type(8)))  _Float16 v8h;
typedef __attribute__((ext_vector_type(8)))  float    v8f;
typedef __attribute__((ext_vector_type(4)))  int      v4i;

#if defined(__has_builtin)
#if __has_builtin(__builtin_amdgcn_global_load_async_to_lds_b128)
#define HAVE_ASYNC_LDS 1
#endif
#endif
#ifndef HAVE_ASYNC_LDS
#define HAVE_ASYNC_LDS 0
#endif

union AF { v16h v; v8h h[2]; _Float16 e[16]; };

__device__ __forceinline__ v8f wmma_f16(v16h a, v16h b, v8f c) {
  // D = A(16x32 f16) * B(32x16 f16) + C(16x16 f32)
  return __builtin_amdgcn_wmma_f32_16x16x32_f16(false, a, false, b, (short)0, c, false, false);
}

// ---------------- f32 -> f16 weight conversion ----------------
__global__ void cvt_kernel(const float* __restrict__ in, _Float16* __restrict__ out, int n) {
  int i = blockIdx.x * blockDim.x + threadIdx.x;
  if (i < n) out[i] = (_Float16)in[i];
}

// ---------------- LayerNorm: one wave per token row ----------------
__global__ void __launch_bounds__(256) ln_kernel(
    const float* __restrict__ x, const float* __restrict__ g,
    const float* __restrict__ b, _Float16* __restrict__ out)
{
  const int wv = threadIdx.x >> 5, lane = threadIdx.x & 31;
  const int row = blockIdx.x * 8 + wv;
  if (row >= NTOK) return;
  const float* xr = x + (size_t)row * DIM;
  float s = 0.f, s2 = 0.f;
#pragma unroll
  for (int i = 0; i < DIM / 32; ++i) {
    float vv = xr[lane + 32 * i];
    s += vv; s2 += vv * vv;
  }
#pragma unroll
  for (int ofs = 16; ofs >= 1; ofs >>= 1) {
    s  += __shfl_xor(s,  ofs);
    s2 += __shfl_xor(s2, ofs);
  }
  float mean = s * (1.0f / DIM);
  float var  = s2 * (1.0f / DIM) - mean * mean;
  float rs   = rsqrtf(var + 1e-5f);
#pragma unroll
  for (int i = 0; i < DIM / 32; ++i) {
    int j = lane + 32 * i;
    out[(size_t)row * DIM + j] = (_Float16)((xr[j] - mean) * rs * g[j] + b[j]);
  }
}

// ---- cooperative weight-slab staging: 32 K-rows x 64 N-cols of f16 (4 KB) ----
// 256 threads: thread t loads one 16-byte chunk (row = t>>3, chunk = t&7).
__device__ __forceinline__ void stage_w(const _Float16* __restrict__ W, int N,
                                        int k0, int n0, _Float16* buf, int tid)
{
  const int row = tid >> 3, chunk = tid & 7;
  const _Float16* g = W + (size_t)(k0 + row) * N + n0 + chunk * 8;
  _Float16* l = buf + row * 64 + chunk * 8;
#if HAVE_ASYNC_LDS
  // CDNA5 async copy: memory -> LDS, no VGPR round trip, tracked by ASYNCcnt
  __builtin_amdgcn_global_load_async_to_lds_b128((v4i*)g, (v4i*)l, 0, 0);
#else
  *(v8h*)l = *(const v8h*)g;
#endif
}

// ---------------- Cooperative WMMA GEMM ----------------
// Block = 8 waves sharing one 64-col slab of W (double-buffered in LDS via async
// loads). Each wave computes a 32x64 tile (8 WMMAs / 32-K step).
// EPI 0: +bias -> f16   EPI 1: +bias, GELU(erf) -> f16   EPI 2: +bias +residual -> f32
template<int EPI>
__global__ void __launch_bounds__(256) gemm_kernel(
    const _Float16* __restrict__ A, const _Float16* __restrict__ W,
    const float* __restrict__ bias, const float* __restrict__ res,
    _Float16* __restrict__ out16, float* __restrict__ out32,
    int M, int N, int K)
{
  __shared__ _Float16 wbuf[2][32 * 64];
  const int tid = threadIdx.x;
  const int wv = tid >> 5, lane = tid & 31;
  const int msuper = M >> 8;
  const int bm = blockIdx.x % msuper;
  const int n0 = (blockIdx.x / msuper) << 6;
  const int m0 = (bm << 8) + (wv << 5);        // 32 rows per wave
  const int ml = lane & 15, hi = lane >> 4;

  const _Float16* arow0 = A + (size_t)(m0 + ml) * K + 8 * hi;        // rows m0..m0+15
  const _Float16* arow1 = A + (size_t)(m0 + 16 + ml) * K + 8 * hi;   // rows m0+16..m0+31
  v8f acc0[4] = {}, acc1[4] = {};

  const int nsteps = K >> 5;
  stage_w(W, N, 0, n0, wbuf[0], tid);
  for (int s = 0; s < nsteps; ++s) {
    const int k0 = s << 5;
    if (s + 1 < nsteps) stage_w(W, N, k0 + 32, n0, wbuf[(s + 1) & 1], tid);
#if HAVE_ASYNC_LDS
    if (s + 1 < nsteps) asm volatile("s_wait_asynccnt 0x1" ::: "memory");
    else                asm volatile("s_wait_asynccnt 0x0" ::: "memory");
#endif
    __syncthreads();                            // slab visible to all 8 waves
    AF a0, a1;
    a0.h[0] = *(const v8h*)(arow0 + k0);
    a0.h[1] = *(const v8h*)(arow0 + k0 + 16);
    a1.h[0] = *(const v8h*)(arow1 + k0);
    a1.h[1] = *(const v8h*)(arow1 + k0 + 16);
    const _Float16* bb = wbuf[s & 1] + ml;   // B-frag: lane%16 = N col of slab, elements = K rows (table order)
#pragma unroll
    for (int t = 0; t < 4; ++t) {
      v16h bf;
#pragma unroll
      for (int e = 0; e < 16; ++e) bf[e] = bb[((e & 7) + 8 * hi + 16 * (e >> 3)) * 64 + 16 * t];
      acc0[t] = wmma_f16(a0.v, bf, acc0[t]);
      acc1[t] = wmma_f16(a1.v, bf, acc1[t]);
    }
    __syncthreads();                            // reads done before re-staging
  }

#pragma unroll
  for (int t = 0; t < 4; ++t) {
#pragma unroll
    for (int r = 0; r < 8; ++r) {
      const int col = n0 + 16 * t + ml;
      const float bz = bias[col];
#pragma unroll
      for (int mm = 0; mm < 2; ++mm) {
        const int row = m0 + 16 * mm + r + 8 * hi;
        float vv = (mm ? acc1[t][r] : acc0[t][r]) + bz;
        if (EPI == 0) {
          out16[(size_t)row * N + col] = (_Float16)vv;
        } else if (EPI == 1) {
          float gl = 0.5f * vv * (1.0f + erff(vv * 0.70710678118654752f));
          out16[(size_t)row * N + col] = (_Float16)gl;
        } else {
          out32[(size_t)row * N + col] = vv + res[(size_t)row * N + col];
        }
      }
    }
  }
}

// ---------------- RoPE + scatter: qkv f16 -> q (scaled, [B,S,H,HD]), kT ([B,H,HD,S]), v ----------------
__global__ void rope_kernel(const _Float16* __restrict__ qkv,
                            const float* __restrict__ cs, const float* __restrict__ sn,
                            _Float16* __restrict__ q, _Float16* __restrict__ kT,
                            _Float16* __restrict__ v)
{
  int idx = blockIdx.x * blockDim.x + threadIdx.x;        // ((b*S+s)*H+h)*HD+d
  if (idx >= NTOK * DIM) return;
  int d = idx & 63;
  int h = (idx >> 6) & 7;
  int s = (idx >> 9) & (SEQ - 1);
  int b = idx >> 20;
  size_t row = (size_t)(b * SEQ + s) * (3 * DIM);
  int col = h * HD + d;
  int dp  = (d < 32) ? d + 32 : d - 32;
  float sgn = (d < 32) ? -1.f : 1.f;
  float qv = (float)qkv[row + col];
  float kv = (float)qkv[row + DIM + col];
  float vv = (float)qkv[row + 2 * DIM + col];
  float qr = (float)qkv[row + h * HD + dp] * sgn;
  float kr = (float)qkv[row + DIM + h * HD + dp] * sgn;
  float c = cs[s * HD + d], si = sn[s * HD + d];
  q[idx] = (_Float16)((qv * c + qr * si) * 0.125f);   // fold 1/sqrt(HD)
  v[idx] = (_Float16)vv;
  kT[((size_t)(b * HEADS + h) * HD + d) * SEQ + s] = (_Float16)(kv * c + kr * si);
}

// ---------------- Flash attention: one wave = 16 query rows of one (b,h) ----------------
__global__ void __launch_bounds__(256) attn_kernel(
    const _Float16* __restrict__ q, const _Float16* __restrict__ kT,
    const _Float16* __restrict__ v, _Float16* __restrict__ o)
{
  __shared__ _Float16 ldsP[8][16 * 32];     // per-wave P tile (16 rows x 32 keys)
  const int wv = threadIdx.x >> 5;
  const int lane = threadIdx.x & 31;
  const int wid = blockIdx.x * 8 + wv;
  if (wid >= BATCH * HEADS * (SEQ / 16)) return;
  const int qt = wid & (SEQ / 16 - 1);
  const int bh = wid / (SEQ / 16);
  const int b = bh >> 3, h = bh & 7;
  const int q0 = qt * 16;
  const int ml = lane & 15, hi = lane >> 4;

  // Q fragments: head dim K=64 -> 2 chunks of 32 (softmax scale folded in)
  const _Float16* qrow = q + ((size_t)(b * SEQ + q0 + ml) * HEADS + h) * HD;
  AF aq[2];
#pragma unroll
  for (int kd = 0; kd < 2; ++kd) {
    aq[kd].h[0] = *(const v8h*)(qrow + kd * 32 + 8 * hi);
    aq[kd].h[1] = *(const v8h*)(qrow + kd * 32 + 16 + 8 * hi);
  }

  float mrow[8], lrow[8];
#pragma unroll
  for (int r = 0; r < 8; ++r) { mrow[r] = -3.0e38f; lrow[r] = 0.f; }
  v8f oacc[4] = {};

  const _Float16* kbase = kT + (size_t)bh * HD * SEQ;
  for (int kc = 0; kc < SEQ; kc += 32) {
    // S tile 16x32 = Q(16x64) * K^T(64x32): 4 WMMAs
    v8f s0 = {}, s1 = {};
#pragma unroll
    for (int kd = 0; kd < 2; ++kd) {
      const _Float16* kp = kbase + (size_t)(kd * 32 + 8 * hi) * SEQ + kc + ml; // lane%16 = key col of B, elements = head-dim rows
      v16h b0, b1;
#pragma unroll
      for (int e = 0; e < 16; ++e) {
        size_t off = (size_t)((e & 7) + 16 * (e >> 3)) * SEQ;
        b0[e] = kp[off];
        b1[e] = kp[off + 16];
      }
      s0 = wmma_f16(aq[kd].v, b0, s0);
      s1 = wmma_f16(aq[kd].v, b1, s1);
    }
    // streaming softmax: per-row stats across the 16-lane half-groups
#pragma unroll
    for (int r = 0; r < 8; ++r) {
      float t = fmaxf(s0[r], s1[r]);
      t = fmaxf(t, __shfl_xor(t, 1));
      t = fmaxf(t, __shfl_xor(t, 2));
      t = fmaxf(t, __shfl_xor(t, 4));
      t = fmaxf(t, __shfl_xor(t, 8));
      float mnew  = fmaxf(mrow[r], t);
      float alpha = __expf(mrow[r] - mnew);
      float p0 = __expf(s0[r] - mnew);
      float p1 = __expf(s1[r] - mnew);
      float rs = p0 + p1;
      rs += __shfl_xor(rs, 1);
      rs += __shfl_xor(rs, 2);
      rs += __shfl_xor(rs, 4);
      rs += __shfl_xor(rs, 8);
      lrow[r] = lrow[r] * alpha + rs;
      mrow[r] = mnew;
#pragma unroll
      for (int t4 = 0; t4 < 4; ++t4) oacc[t4][r] *= alpha;
      int prow = r + 8 * hi;                                 // C-frag row mapping
      ldsP[wv][prow * 32 + ml]      = (_Float16)p0;
      ldsP[wv][prow * 32 + 16 + ml] = (_Float16)p1;
    }
    asm volatile("s_wait_dscnt 0" ::: "memory");             // LDS RAW before A-frag reload
    AF ap;                                                   // P in A-fragment layout
    const _Float16* pr = &ldsP[wv][ml * 32 + 8 * hi];
    ap.h[0] = *(const v8h*)(pr);
    ap.h[1] = *(const v8h*)(pr + 16);
    // O += P(16x32) * V(32x64): 4 WMMAs
#pragma unroll
    for (int t4 = 0; t4 < 4; ++t4) {
      const _Float16* vp = v + ((size_t)(b * SEQ + kc + 8 * hi) * HEADS + h) * HD + 16 * t4 + ml;
      v16h bv;
#pragma unroll
      for (int e = 0; e < 16; ++e) bv[e] = vp[(size_t)((e & 7) + 16 * (e >> 3)) * HEADS * HD];
      oacc[t4] = wmma_f16(ap.v, bv, oacc[t4]);
    }
  }
  // normalize and write o as [B,S,512] f16 (input to proj GEMM)
#pragma unroll
  for (int t4 = 0; t4 < 4; ++t4) {
#pragma unroll
    for (int r = 0; r < 8; ++r) {
      size_t row = (size_t)(b * SEQ + q0 + r + 8 * hi);
      o[row * DIM + h * HD + 16 * t4 + ml] = (_Float16)(oacc[t4][r] / lrow[r]);
    }
  }
}

extern "C" void kernel_launch(void* const* d_in, const int* in_sizes, int n_in,
                              void* d_out, int out_size, void* d_ws, size_t ws_size,
                              hipStream_t stream)
{
  const float* x      = (const float*)d_in[0];
  const float* ropec  = (const float*)d_in[1];
  const float* ropes  = (const float*)d_in[2];
  const float* n1g    = (const float*)d_in[3];
  const float* n1b    = (const float*)d_in[4];
  const float* n2g    = (const float*)d_in[5];
  const float* n2b    = (const float*)d_in[6];
  const float* w_qkv  = (const float*)d_in[7];
  const float* b_qkv  = (const float*)d_in[8];
  const float* w_proj = (const float*)d_in[9];
  const float* b_proj = (const float*)d_in[10];
  const float* w_fc1  = (const float*)d_in[11];
  const float* b_fc1  = (const float*)d_in[12];
  const float* w_fc2  = (const float*)d_in[13];
  const float* b_fc2  = (const float*)d_in[14];
  float* out = (float*)d_out;
  (void)in_sizes; (void)n_in; (void)out_size; (void)ws_size;

  char* p = (char*)d_ws;
  auto alloc = [&](size_t bytes) { char* r = p; p += (bytes + 255) & ~(size_t)255; return r; };
  _Float16* h1   = (_Float16*)alloc((size_t)NTOK * DIM * 2);
  _Float16* wq16 = (_Float16*)alloc((size_t)DIM * 3 * DIM * 2);
  _Float16* qkv  = (_Float16*)alloc((size_t)NTOK * 3 * DIM * 2);
  _Float16* qf   = (_Float16*)alloc((size_t)NTOK * DIM * 2);
  _Float16* kTf  = (_Float16*)alloc((size_t)NTOK * DIM * 2);
  _Float16* vf   = (_Float16*)alloc((size_t)NTOK * DIM * 2);
  _Float16* of   = (_Float16*)alloc((size_t)NTOK * DIM * 2);
  _Float16* wp16 = (_Float16*)alloc((size_t)DIM * DIM * 2);
  float*    x2   = (float*)   alloc((size_t)NTOK * DIM * 4);
  _Float16* h2   = (_Float16*)alloc((size_t)NTOK * DIM * 2);
  _Float16* w116 = (_Float16*)alloc((size_t)DIM * 4 * DIM * 2);
  _Float16* mlp  = (_Float16*)alloc((size_t)NTOK * 4 * DIM * 2);
  _Float16* w216 = (_Float16*)alloc((size_t)4 * DIM * DIM * 2);

  // weights f32 -> f16 (L2-resident thereafter)
  cvt_kernel<<<(DIM * 3 * DIM + 255) / 256, 256, 0, stream>>>(w_qkv,  wq16, DIM * 3 * DIM);
  cvt_kernel<<<(DIM * DIM + 255) / 256,     256, 0, stream>>>(w_proj, wp16, DIM * DIM);
  cvt_kernel<<<(DIM * 4 * DIM + 255) / 256, 256, 0, stream>>>(w_fc1,  w116, DIM * 4 * DIM);
  cvt_kernel<<<(4 * DIM * DIM + 255) / 256, 256, 0, stream>>>(w_fc2,  w216, 4 * DIM * DIM);

  // LN1 -> f16
  ln_kernel<<<NTOK / 8, 256, 0, stream>>>(x, n1g, n1b, h1);
  // QKV GEMM (+bias, f16): grid = (M/256) * (N/64)
  gemm_kernel<0><<<(NTOK / 256) * (3 * DIM / 64), 256, 0, stream>>>(
      h1, wq16, b_qkv, nullptr, qkv, nullptr, NTOK, 3 * DIM, DIM);
  // RoPE + q-scale + K transpose
  rope_kernel<<<(NTOK * DIM) / 256, 256, 0, stream>>>(qkv, ropec, ropes, qf, kTf, vf);
  // flash attention
  attn_kernel<<<(BATCH * HEADS * (SEQ / 16)) / 8, 256, 0, stream>>>(qf, kTf, vf, of);
  // proj GEMM (+bias +residual x, f32)
  gemm_kernel<2><<<(NTOK / 256) * (DIM / 64), 256, 0, stream>>>(
      of, wp16, b_proj, x, nullptr, x2, NTOK, DIM, DIM);
  // LN2 -> f16
  ln_kernel<<<NTOK / 8, 256, 0, stream>>>(x2, n2g, n2b, h2);
  // FC1 GEMM (+bias, exact GELU, f16)
  gemm_kernel<1><<<(NTOK / 256) * (4 * DIM / 64), 256, 0, stream>>>(
      h2, w116, b_fc1, nullptr, mlp, nullptr, NTOK, 4 * DIM, DIM);
  // FC2 GEMM (+bias +residual x2, f32) -> d_out
  gemm_kernel<2><<<(NTOK / 256) * (DIM / 64), 256, 0, stream>>>(
      mlp, w216, b_fc2, x2, nullptr, out, NTOK, DIM, 4 * DIM);
}
